// MCattention_11441792876624
// MI455X (gfx1250) — matrix operand order repaired and checked by reading, not compiled, not run on hardware
//
#include <hip/hip_runtime.h>
#include <hip/hip_bf16.h>
#include <math.h>

// ---------------- problem constants ----------------
#define L_T   20736          // H*W = 144*144
#define C_T   384
#define HW_   144
#define NH_   4
#define D_    96
#define SCALE_ 0.10206207261596575f  // 96^-0.5

// ---------------- WMMA types ----------------
typedef _Float16 v16h __attribute__((ext_vector_type(16)));
typedef _Float16 h8   __attribute__((ext_vector_type(8)));
typedef float    v8f  __attribute__((ext_vector_type(8)));

__device__ __forceinline__ v16h ld_frag(const _Float16* p, int second_off) {
    union { v16h v; h8 h[2]; } u;
    u.h[0] = *(const h8*)(p);
    u.h[1] = *(const h8*)(p + second_off);
    return u.v;
}

__device__ __forceinline__ float wred(float v) {
    #pragma unroll
    for (int off = 16; off > 0; off >>= 1) v += __shfl_xor(v, off, 32);
    return v;
}

// CDNA5 async global->LDS copy, 16 bytes per lane (ASYNCcnt-tracked DMA).
// VDST = 32-bit LDS byte offset, VADDR = 64-bit global address.
__device__ __forceinline__ void async_cp16(const _Float16* gsrc, const _Float16* ldst) {
    unsigned lds_off = (unsigned)(size_t)ldst;     // low 32 bits of generic = LDS offset
    asm volatile("global_load_async_to_lds_b128 %0, %1, off"
                 :: "v"(lds_off), "v"(gsrc) : "memory");
}
// async loads complete in order: waiting <=3 releases the previous 3-op tile
__device__ __forceinline__ void async_wait0() {
    asm volatile("s_wait_asynccnt 0x0" ::: "memory");
}
__device__ __forceinline__ void async_wait3() {
    asm volatile("s_wait_asynccnt 0x3" ::: "memory");
}

// ---------------- f32 -> f16 bulk convert (vectorized) ----------------
__global__ __launch_bounds__(256) void k_cvt_f16(const float4* __restrict__ X,
                                                 _Float16* __restrict__ Y, int n4) {
    const int i = blockIdx.x * 256 + threadIdx.x;
    if (i >= n4) return;
    const float4 v = X[i];
    union { _Float16 h[4]; unsigned long long u; } o;
    o.h[0] = (_Float16)v.x; o.h[1] = (_Float16)v.y;
    o.h[2] = (_Float16)v.z; o.h[3] = (_Float16)v.w;
    *(unsigned long long*)(Y + (size_t)i * 4) = o.u;
}

// ---------------- transpose (C,L) -> (L,C) ----------------
__global__ __launch_bounds__(256) void k_transpose(const float* __restrict__ X,
                                                   float* __restrict__ Y) {
    __shared__ float tile[32][33];
    const int tx = threadIdx.x, ty = threadIdx.y;      // blockDim (32,8)
    const int l0 = blockIdx.x * 32, c0 = blockIdx.y * 32;
    #pragma unroll
    for (int j = 0; j < 4; ++j)
        tile[ty + 8 * j][tx] = X[(size_t)(c0 + ty + 8 * j) * L_T + l0 + tx];
    __syncthreads();
    #pragma unroll
    for (int j = 0; j < 4; ++j)
        Y[(size_t)(l0 + ty + 8 * j) * C_T + c0 + tx] = tile[tx][ty + 8 * j];
}

// ---------------- LayerNorm: one wave per token ----------------
__global__ __launch_bounds__(256) void k_layernorm(const float* __restrict__ X,
                                                   const float* __restrict__ g,
                                                   const float* __restrict__ b,
                                                   float* __restrict__ Y) {
    const int lane = threadIdx.x & 31;
    const int wave = threadIdx.x >> 5;
    const size_t l = (size_t)blockIdx.x * 8 + wave;
    const float* xr = X + l * C_T;
    float v[12];
    float s = 0.f;
    #pragma unroll
    for (int i = 0; i < 12; ++i) { v[i] = xr[lane + 32 * i]; s += v[i]; }
    s = wred(s);
    const float mean = s * (1.f / 384.f);
    float sq = 0.f;
    #pragma unroll
    for (int i = 0; i < 12; ++i) { float d = v[i] - mean; sq += d * d; }
    sq = wred(sq);
    const float rstd = rsqrtf(sq * (1.f / 384.f) + 1e-5f);
    float* yr = Y + l * C_T;
    #pragma unroll
    for (int i = 0; i < 12; ++i) {
        int c = lane + 32 * i;
        yr[c] = (v[i] - mean) * rstd * g[c] + b[c];
    }
}

// ---- WMMA GEMM (f16 in, f32 acc): Out[M,N] = act(A[M,K]*W[N,K]^T + bias) (+resid) ----
#define BM 128
#define BN 64
#define BK 32
#define LDT 40   // padded LDS stride in halves (80B rows, 16B-aligned, conflict-free)

__device__ __forceinline__ void stage_tile(const _Float16* __restrict__ A,
                                           const _Float16* __restrict__ W,
                                           _Float16* As, _Float16* Bs,
                                           int m0, int n0, int k0, int K, int tid) {
    int id = tid;                              // A chunk 0..255
    int r = id >> 2, c = (id & 3) * 8;
    async_cp16(A + (size_t)(m0 + r) * K + k0 + c, &As[r * LDT + c]);
    id = tid + 256;                            // A chunk 256..511
    r = id >> 2; c = (id & 3) * 8;
    async_cp16(A + (size_t)(m0 + r) * K + k0 + c, &As[r * LDT + c]);
    id = tid;                                  // B chunk 0..255
    r = id >> 2; c = (id & 3) * 8;
    async_cp16(W + (size_t)(n0 + r) * K + k0 + c, &Bs[r * LDT + c]);
}

__global__ __launch_bounds__(256) void k_gemm_wmma(const _Float16* __restrict__ A,
                                                   const _Float16* __restrict__ W,
                                                   float* __restrict__ Out,
                                                   const float* __restrict__ bias,
                                                   const float* __restrict__ resid,
                                                   int M, int N, int K, int gelu) {
    __shared__ __attribute__((aligned(16))) _Float16 As[2][BM * LDT];
    __shared__ __attribute__((aligned(16))) _Float16 Bs[2][BN * LDT];
    const int tid  = threadIdx.x;
    const int lane = tid & 31;
    const int wave = tid >> 5;
    const int wm = wave >> 1;          // 0..3 -> 32-row strip
    const int wn = wave & 1;           // 0..1 -> 32-col strip
    const int m0 = blockIdx.x * BM;
    const int n0 = blockIdx.y * BN;

    v8f acc00 = {}, acc01 = {}, acc10 = {}, acc11 = {};

    const int nk = K / BK;
    stage_tile(A, W, As[0], Bs[0], m0, n0, 0, K, tid);   // prefetch tile 0

    for (int kt = 0; kt < nk; ++kt) {
        const int buf = kt & 1;
        if (kt + 1 < nk) {
            // issue next tile's DMA into the other buffer, then release only
            // the current tile's 3 in-order async ops
            stage_tile(A, W, As[buf ^ 1], Bs[buf ^ 1], m0, n0, (kt + 1) * BK, K, tid);
            async_wait3();
        } else {
            async_wait0();
        }
        __syncthreads();

        // fragment loads per ISA 7.12.2 layouts
        const int rowA = wm * 32 + (lane & 15);
        const int kA   = (lane >> 4) * 8;         // lanes 16-31: K starts at 8 (then +16)
        const int colB = wn * 32 + (lane & 15);
        const int kB   = (lane >> 4) * 8;         // lanes 16-31: K starts at 8 (then +16)
        v16h a0 = ld_frag(&As[buf][rowA * LDT + kA], 16);
        v16h a1 = ld_frag(&As[buf][(rowA + 16) * LDT + kA], 16);
        v16h b0 = ld_frag(&Bs[buf][colB * LDT + kB], 16);
        v16h b1 = ld_frag(&Bs[buf][(colB + 16) * LDT + kB], 16);

        acc00 = __builtin_amdgcn_wmma_f32_16x16x32_f16(false, a0, false, b0, (short)0, acc00, false, false);
        acc01 = __builtin_amdgcn_wmma_f32_16x16x32_f16(false, a0, false, b1, (short)0, acc01, false, false);
        acc10 = __builtin_amdgcn_wmma_f32_16x16x32_f16(false, a1, false, b0, (short)0, acc10, false, false);
        acc11 = __builtin_amdgcn_wmma_f32_16x16x32_f16(false, a1, false, b1, (short)0, acc11, false, false);
        __syncthreads();   // all waves done reading buf before it is re-filled
    }

    // epilogue: C/D layout — col = lane&15, rows = (lane>=16?8:0)+r
    const int rowBase = (lane >> 4) * 8;
    const int colIn   = lane & 15;
    #pragma unroll
    for (int tm = 0; tm < 2; ++tm) {
        #pragma unroll
        for (int tn = 0; tn < 2; ++tn) {
            v8f acc = (tm == 0) ? (tn == 0 ? acc00 : acc01)
                                : (tn == 0 ? acc10 : acc11);
            const int colg = n0 + wn * 32 + tn * 16 + colIn;
            const float bsv = bias ? bias[colg] : 0.f;
            #pragma unroll
            for (int r = 0; r < 8; ++r) {
                const int rowg = m0 + wm * 32 + tm * 16 + rowBase + r;
                float v = acc[r] + bsv;
                if (gelu) v = 0.5f * v * (1.f + erff(v * 0.70710678118654752f));
                if (resid) v += resid[(size_t)rowg * N + colg];
                Out[(size_t)rowg * N + colg] = v;
            }
        }
    }
}

// ---------------- local (5-key windowed) attention + LePE ----------------
__global__ __launch_bounds__(256) void k_attn_local(const float* __restrict__ qkv,
                                                    const float* __restrict__ wv,
                                                    const float* __restrict__ bv,
                                                    float* __restrict__ aout,
                                                    int dirIdx, int accum) {
    const int lane = threadIdx.x & 31;
    const int wave = threadIdx.x >> 5;
    const int wg = blockIdx.x * 8 + wave;
    const int stripe = wg / HW_;
    const int s = wg - stripe * HW_;
    auto lpos = [&](int t) { return dirIdx == 0 ? stripe * HW_ + t : t * HW_ + stripe; };
    const int l = lpos(s);

    for (int h = 0; h < NH_; ++h) {
        const int cb = h * D_;
        const size_t qb = (size_t)l * 1152 + cb;
        const float q0 = qkv[qb + lane]      * SCALE_;
        const float q1 = qkv[qb + lane + 32] * SCALE_;
        const float q2 = qkv[qb + lane + 64] * SCALE_;

        float m = -1e30f, ssum = 0.f, a0 = 0.f, a1 = 0.f, a2 = 0.f;
        #pragma unroll
        for (int j = 0; j < 5; ++j) {
            const int sp = s - 2 + j;
            float logit, v0, v1, v2;
            if (sp >= 0 && sp < HW_) {
                const size_t lb = (size_t)lpos(sp) * 1152 + cb;
                const float k0 = qkv[lb + 384 + lane];
                const float k1 = qkv[lb + 384 + lane + 32];
                const float k2 = qkv[lb + 384 + lane + 64];
                v0 = qkv[lb + 768 + lane];
                v1 = qkv[lb + 768 + lane + 32];
                v2 = qkv[lb + 768 + lane + 64];
                logit = wred(q0 * k0 + q1 * k1 + q2 * k2);
            } else {                     // zero-padded key: logit 0, value 0
                logit = 0.f; v0 = v1 = v2 = 0.f;
            }
            const float mn = fmaxf(m, logit);
            const float corr = __expf(m - mn);
            const float p = __expf(logit - mn);
            ssum = ssum * corr + p;
            a0 = a0 * corr + p * v0;
            a1 = a1 * corr + p * v1;
            a2 = a2 * corr + p * v2;
            m = mn;
        }
        float e0 = bv[cb + lane], e1 = bv[cb + lane + 32], e2 = bv[cb + lane + 64];
        #pragma unroll
        for (int t = 0; t < 3; ++t) {
            const int sp = s - 1 + t;
            if (sp < 0 || sp >= HW_) continue;
            const size_t lb = (size_t)lpos(sp) * 1152 + 768 + cb;
            const int off = (dirIdx == 0) ? (3 + t) : (t * 3 + 1);
            e0 += wv[(cb + lane)      * 9 + off] * qkv[lb + lane];
            e1 += wv[(cb + lane + 32) * 9 + off] * qkv[lb + lane + 32];
            e2 += wv[(cb + lane + 64) * 9 + off] * qkv[lb + lane + 64];
        }
        const float inv = 1.f / ssum;
        const size_t ob = (size_t)l * C_T + cb;
        const float o0 = a0 * inv + e0, o1 = a1 * inv + e1, o2 = a2 * inv + e2;
        if (accum) {
            aout[ob + lane] += o0; aout[ob + lane + 32] += o1; aout[ob + lane + 64] += o2;
        } else {
            aout[ob + lane] = o0; aout[ob + lane + 32] = o1; aout[ob + lane + 64] = o2;
        }
    }
}

// ---------------- atrous (dilation-3, 48-key) attention + LePE ----------------
__global__ __launch_bounds__(256) void k_attn_atrous(const float* __restrict__ qkv,
                                                     const float* __restrict__ wv,
                                                     const float* __restrict__ bv,
                                                     float* __restrict__ aout,
                                                     int dirIdx, int accum) {
    const int lane = threadIdx.x & 31;
    const int wave = threadIdx.x >> 5;
    const int wg = blockIdx.x * 8 + wave;
    const int stripe = wg / HW_;
    const int s = wg - stripe * HW_;
    auto lpos = [&](int t) { return dirIdx == 0 ? stripe * HW_ + t : t * HW_ + stripe; };
    const int l = lpos(s);
    const int j0 = s % 3;

    for (int h = 0; h < NH_; ++h) {
        const int cb = h * D_;
        const size_t qb = (size_t)l * 1152 + cb;
        const float q0 = qkv[qb + lane]      * SCALE_;
        const float q1 = qkv[qb + lane + 32] * SCALE_;
        const float q2 = qkv[qb + lane + 64] * SCALE_;

        float m = -1e30f, ssum = 0.f, a0 = 0.f, a1 = 0.f, a2 = 0.f;
        for (int i = 0; i < 48; ++i) {          // keys with p' ≡ s (mod 3)
            const int sp = j0 + 3 * i;
            const size_t lb = (size_t)lpos(sp) * 1152 + cb;
            const float k0 = qkv[lb + 384 + lane];
            const float k1 = qkv[lb + 384 + lane + 32];
            const float k2 = qkv[lb + 384 + lane + 64];
            const float logit = wred(q0 * k0 + q1 * k1 + q2 * k2);
            const float mn = fmaxf(m, logit);
            const float corr = __expf(m - mn);
            const float p = __expf(logit - mn);
            ssum = ssum * corr + p;
            a0 = a0 * corr + p * qkv[lb + 768 + lane];
            a1 = a1 * corr + p * qkv[lb + 768 + lane + 32];
            a2 = a2 * corr + p * qkv[lb + 768 + lane + 64];
            m = mn;
        }
        float e0 = bv[cb + lane], e1 = bv[cb + lane + 32], e2 = bv[cb + lane + 64];
        #pragma unroll
        for (int t = 0; t < 3; ++t) {
            const int sp = s - 1 + t;
            if (sp < 0 || sp >= HW_) continue;
            const size_t lb = (size_t)lpos(sp) * 1152 + 768 + cb;
            const int off = (dirIdx == 0) ? (3 + t) : (t * 3 + 1);
            e0 += wv[(cb + lane)      * 9 + off] * qkv[lb + lane];
            e1 += wv[(cb + lane + 32) * 9 + off] * qkv[lb + lane + 32];
            e2 += wv[(cb + lane + 64) * 9 + off] * qkv[lb + lane + 64];
        }
        const float inv = 1.f / ssum;
        const size_t ob = (size_t)l * C_T + cb;
        const float o0 = a0 * inv + e0, o1 = a1 * inv + e1, o2 = a2 * inv + e2;
        if (accum) {
            aout[ob + lane] += o0; aout[ob + lane + 32] += o1; aout[ob + lane + 64] += o2;
        } else {
            aout[ob + lane] = o0; aout[ob + lane + 32] = o1; aout[ob + lane + 64] = o2;
        }
    }
}

// ---------------- host launcher ----------------
extern "C" void kernel_launch(void* const* d_in, const int* in_sizes, int n_in,
                              void* d_out, int out_size, void* d_ws, size_t ws_size,
                              hipStream_t stream) {
    (void)in_sizes; (void)n_in; (void)out_size; (void)ws_size;
    const float* x      = (const float*)d_in[0];
    const float* g1     = (const float*)d_in[1];
    const float* b1     = (const float*)d_in[2];
    const float* w_qkv  = (const float*)d_in[3];
    const float* w_proj = (const float*)d_in[4];
    const float* b_proj = (const float*)d_in[5];
    const float* g2     = (const float*)d_in[6];
    const float* b2     = (const float*)d_in[7];
    const float* w_mlp1 = (const float*)d_in[8];
    const float* b_mlp1 = (const float*)d_in[9];
    const float* w_mlp2 = (const float*)d_in[10];
    const float* b_mlp2 = (const float*)d_in[11];
    const float* g3     = (const float*)d_in[12];
    const float* b3     = (const float*)d_in[13];
    const float* wv_hL  = (const float*)d_in[14];
    const float* bv_hL  = (const float*)d_in[15];
    const float* wv_vL  = (const float*)d_in[16];
    const float* bv_vL  = (const float*)d_in[17];
    const float* wv_hA  = (const float*)d_in[18];
    const float* bv_hA  = (const float*)d_in[19];
    const float* wv_vA  = (const float*)d_in[20];
    const float* bv_vA  = (const float*)d_in[21];
    float* out = (float*)d_out;

    const size_t LC = (size_t)L_T * C_T;
    float* s_x   = (float*)d_ws;             // residual stream
    float* s_ln  = s_x + LC;                 // LN output / MLP hidden
    float* s_a   = s_ln + LC;                // attention output
    float* s_qkv = s_a + LC;                 // 3*LC
    _Float16* a16 = (_Float16*)(s_qkv + 3 * LC);   // f16 mirror of A
    _Float16* w16 = a16 + LC;                      // f16 mirror of W (<= 3C*C)

    const dim3 tpbT(32, 8);
    const int LNB = L_T / 8;                 // 2592 blocks, 8 tokens each
    const dim3 gQKV(L_T / BM, 1152 / BN);    // 162 x 18
    const dim3 gSQ (L_T / BM,  384 / BN);    // 162 x 6

    auto cvt = [&](const float* src, _Float16* dst, size_t n) {
        k_cvt_f16<<<(unsigned)(n / 4 / 256), 256, 0, stream>>>((const float4*)src, dst, (int)(n / 4));
    };

    // block 1: local CSWin attention
    k_transpose <<<dim3(L_T / 32, C_T / 32), tpbT, 0, stream>>>(x, s_x);
    k_layernorm <<<LNB, 256, 0, stream>>>(s_x, g1, b1, s_ln);
    cvt(s_ln, a16, LC);  cvt(w_qkv, w16, (size_t)1152 * 384);
    k_gemm_wmma <<<gQKV, 256, 0, stream>>>(a16, w16, s_qkv, nullptr, nullptr, L_T, 1152, 384, 0);
    k_attn_local<<<LNB, 256, 0, stream>>>(s_qkv, wv_hL, bv_hL, s_a, 0, 0);
    k_attn_local<<<LNB, 256, 0, stream>>>(s_qkv, wv_vL, bv_vL, s_a, 1, 1);
    cvt(s_a, a16, LC);   cvt(w_proj, w16, (size_t)384 * 384);
    k_gemm_wmma <<<gSQ, 256, 0, stream>>>(a16, w16, s_x, b_proj, s_x, L_T, 384, 384, 0);
    k_layernorm <<<LNB, 256, 0, stream>>>(s_x, g2, b2, s_x);

    // block 2: atrous attention
    cvt(s_x, a16, LC);   cvt(w_qkv, w16, (size_t)1152 * 384);
    k_gemm_wmma  <<<gQKV, 256, 0, stream>>>(a16, w16, s_qkv, nullptr, nullptr, L_T, 1152, 384, 0);
    k_attn_atrous<<<LNB, 256, 0, stream>>>(s_qkv, wv_hA, bv_hA, s_a, 0, 0);
    k_attn_atrous<<<LNB, 256, 0, stream>>>(s_qkv, wv_vA, bv_vA, s_a, 1, 1);
    cvt(s_a, a16, LC);   cvt(w_proj, w16, (size_t)384 * 384);
    k_gemm_wmma  <<<gSQ, 256, 0, stream>>>(a16, w16, s_x, b_proj, s_x, L_T, 384, 384, 0);
    k_layernorm  <<<LNB, 256, 0, stream>>>(s_x, g2, b2, s_x);

    // MLP + final LN
    cvt(s_x, a16, LC);   cvt(w_mlp1, w16, (size_t)384 * 384);
    k_gemm_wmma <<<gSQ, 256, 0, stream>>>(a16, w16, s_ln, b_mlp1, nullptr, L_T, 384, 384, 1);
    cvt(s_ln, a16, LC);  cvt(w_mlp2, w16, (size_t)384 * 384);
    k_gemm_wmma <<<gSQ, 256, 0, stream>>>(a16, w16, s_x, b_mlp2, s_x, L_T, 384, 384, 0);
    k_layernorm <<<LNB, 256, 0, stream>>>(s_x, g3, b3, out);
}
